// MAGICGen_VLMAG_65146063946602
// MI455X (gfx1250) — compile-verified, matrix operand order checked by reading, not run on hardware
//
#include <hip/hip_runtime.h>

typedef __attribute__((ext_vector_type(16))) __bf16 v16bf;
typedef __attribute__((ext_vector_type(8)))  float  v8f;

#define B_SZ     4096
#define HIDN     32
#define NLAY     6
#define VOCAB_N  361
#define VOCAB_P  368
#define TSTEPS   24
#define KCURVES  8

struct __align__(16) U128 { unsigned int w0, w1, w2, w3; };
union BF16x16 { U128 q[2]; v16bf v; unsigned short u[16]; };

__device__ __forceinline__ unsigned short f2bf(float f) {
  union { float f; unsigned int u; } x; x.f = f;
  unsigned int u = x.u;
  u += 0x7FFFu + ((u >> 16) & 1u);
  return (unsigned short)(u >> 16);
}

__device__ __forceinline__ v8f wmma_bf16(v16bf a, v16bf b, v8f c) {
  // v_wmma_f32_16x16x32_bf16 : D = A(16x32) * B(32x16) + C(16x16 f32)
  return __builtin_amdgcn_wmma_f32_16x16x32_bf16(false, a, false, b, (short)0, c,
                                                 false, false);
}

__device__ __forceinline__ float sigmoidf_fast(float x) {
  return 1.0f / (1.0f + __expf(-x));
}

// gfx1250 async copy: global -> LDS, 16 bytes per active lane, ASYNCcnt-tracked.
// lds_off is the wave-relative LDS byte offset (low 32 bits of generic pointer).
__device__ __forceinline__ void async_load_b128(unsigned lds_off, const void* gptr) {
  asm volatile("global_load_async_to_lds_b128 %0, %1, off"
               :: "v"(lds_off), "v"((unsigned long long)(size_t)gptr)
               : "memory");
}
__device__ __forceinline__ void wait_async0() {
  asm volatile("s_wait_asynccnt 0x0" ::: "memory");
}

// ---------------------------------------------------------------------------
// Weight prep: f32 (L, K, N) row-major -> bf16 (L, Npad, K) "transposed",
// zero-padding columns n >= N.
// ---------------------------------------------------------------------------
__global__ void k_transpose_bf16(const float* __restrict__ src,
                                 unsigned short* __restrict__ dst,
                                 int K, int N, int Npad, int L) {
  long total = (long)L * Npad * K;
  for (long idx = (long)blockIdx.x * blockDim.x + threadIdx.x; idx < total;
       idx += (long)gridDim.x * blockDim.x) {
    int k = (int)(idx % K);
    long r = idx / K;
    int n = (int)(r % Npad);
    int l = (int)(r / Npad);
    float v = (n < N) ? src[(size_t)l * K * N + (size_t)k * N + n] : 0.0f;
    dst[idx] = f2bf(v);
  }
}

// Re-initialize LSTM state every launch (deterministic across graph replays).
__global__ void k_init_state(const float* __restrict__ hidden,
                             const float* __restrict__ cell,
                             unsigned short* __restrict__ Hbuf,
                             float* __restrict__ Cbuf, int n) {
  for (int i = blockIdx.x * blockDim.x + threadIdx.x; i < n;
       i += gridDim.x * blockDim.x) {
    Hbuf[i] = f2bf(hidden[i]);
    Cbuf[i] = cell[i];
  }
}

// ---------------------------------------------------------------------------
// Fused particle encoder: per batch row b,
//   x slab (100x64 f32) staged into LDS via async b128 copies,
//   act1 = relu(x_p @ W1 + b1)   (100x64 -> 100x256), padded to 112 rows, LDS
//   act2 = relu(act1 @ W2 + b2)  (100x256 -> 100x256)
//   mean over 100 particles -> A3[b, 64:320]; A3[b, 0:64] = bf16(c[b])
// One block (8 waves) per batch row; wave-per-16x16 WMMA tiles.
// Dynamic LDS layout: xslab f32[112*64] | act1 bf16[112*256] | colsum f32[256]
// ---------------------------------------------------------------------------
#define ENC_LDS_BYTES (112 * 64 * 4 + 112 * 256 * 2 + 256 * 4)  // 87040

__global__ void __launch_bounds__(256)
k_particle_enc(const float* __restrict__ x, const float* __restrict__ cvec,
               const unsigned short* __restrict__ W1t, const float* __restrict__ b1,
               const unsigned short* __restrict__ W2t, const float* __restrict__ b2,
               unsigned short* __restrict__ A3) {
  extern __shared__ char smem[];
  float* xslab = (float*)smem;                                        // 28672 B
  unsigned short* act1 = (unsigned short*)(smem + 112 * 64 * 4);      // 57344 B
  float* colsum = (float*)(smem + 112 * 64 * 4 + 112 * 256 * 2);      // 1024 B

  const int b = blockIdx.x;
  const int tid = threadIdx.x;
  const int wave = tid >> 5, lane = tid & 31;
  const int lrow = lane & 15, lhalf = lane >> 4;

  colsum[tid] = 0.0f;

  // ---- async-stage this block's particle slab: 100*64 f32 = 1600 x b128 --
  {
    const float* xb = x + (size_t)b * 6400;
#pragma unroll
    for (int i = 0; i < 7; ++i) {
      const int chunk = tid + 256 * i;  // one 16B chunk per lane
      if (chunk < 1600) {
        const unsigned lds_off = (unsigned)(size_t)(const void*)(xslab + chunk * 4);
        async_load_b128(lds_off, xb + chunk * 4);
      }
    }
  }
  wait_async0();
  __syncthreads();

  // ---- fc1: 7 M-tiles x 16 N-tiles = 112 tiles, 14 per wave --------------
  for (int it = 0; it < 14; ++it) {
    const int tile = wave + 8 * it;       // 0..111
    const int mt = tile >> 4;             // 0..6
    const int nt = tile & 15;             // 0..15
    const int row = mt * 16 + lrow;       // particle index (padded)
    v8f acc = {};
#pragma unroll
    for (int kc = 0; kc < 2; ++kc) {
      BF16x16 a, bw;
      if (row < 100) {
        const float* xp = xslab + row * 64 + kc * 32 + lhalf * 16;
#pragma unroll
        for (int i = 0; i < 16; ++i) a.u[i] = f2bf(xp[i]);
      } else {
#pragma unroll
        for (int i = 0; i < 16; ++i) a.u[i] = 0;
      }
      const int n = nt * 16 + lrow;
      const U128* wp = (const U128*)(W1t + (size_t)n * 64 + kc * 32 + lhalf * 16);
      bw.q[0] = wp[0]; bw.q[1] = wp[1];
      acc = wmma_bf16(a.v, bw.v, acc);
    }
    const int col = nt * 16 + lrow;
    const float bias = b1[col];
#pragma unroll
    for (int v = 0; v < 8; ++v) {
      const int r = mt * 16 + v + 8 * lhalf;
      act1[r * 256 + col] = f2bf(fmaxf(acc[v] + bias, 0.0f));
    }
  }
  __syncthreads();

  // ---- fc2 + masked particle-mean reduction ------------------------------
  for (int it = 0; it < 14; ++it) {
    const int tile = wave + 8 * it;
    const int mt = tile >> 4;
    const int nt = tile & 15;
    const int arow = mt * 16 + lrow;
    v8f acc = {};
#pragma unroll
    for (int kc = 0; kc < 8; ++kc) {
      BF16x16 a, bw;
      const U128* ap = (const U128*)(act1 + arow * 256 + kc * 32 + lhalf * 16);
      a.q[0] = ap[0]; a.q[1] = ap[1];
      const int n = nt * 16 + lrow;
      const U128* wp = (const U128*)(W2t + (size_t)n * 256 + kc * 32 + lhalf * 16);
      bw.q[0] = wp[0]; bw.q[1] = wp[1];
      acc = wmma_bf16(a.v, bw.v, acc);
    }
    const int col = nt * 16 + lrow;
    const float bias = b2[col];
#pragma unroll
    for (int v = 0; v < 8; ++v) {
      const int r = mt * 16 + v + 8 * lhalf;
      if (r < 100) {
        atomicAdd(&colsum[col], fmaxf(acc[v] + bias, 0.0f));
      }
    }
  }
  __syncthreads();

  if (tid < 64) A3[(size_t)b * 320 + tid] = f2bf(cvec[(size_t)b * 64 + tid]);
  A3[(size_t)b * 320 + 64 + tid] = f2bf(colsum[tid] * (1.0f / 100.0f));
}

// ---------------------------------------------------------------------------
// Generic bf16 WMMA GEMM: out(MxN) = act(A(MxK) @ Wt^T + bias), out in bf16.
// Wt is (N x K) bf16 (column weights contiguous). One 16x16 tile per wave.
// ---------------------------------------------------------------------------
__global__ void __launch_bounds__(256)
k_gemm(const unsigned short* __restrict__ A, const unsigned short* __restrict__ Wt,
       const float* __restrict__ bias, unsigned short* __restrict__ out,
       int M, int K, int N, int do_relu) {
  const int wave = threadIdx.x >> 5, lane = threadIdx.x & 31;
  const int lrow = lane & 15, lhalf = lane >> 4;
  const int ntN = N >> 4;
  const int tile = blockIdx.x * 8 + wave;
  if (tile >= (M >> 4) * ntN) return;
  const int mt = tile / ntN, nt = tile % ntN;
  const int row = mt * 16 + lrow;
  const int n = nt * 16 + lrow;

  v8f acc = {};
  for (int kc = 0; kc < K; kc += 32) {
    BF16x16 a, bw;
    const U128* ap = (const U128*)(A + (size_t)row * K + kc + lhalf * 16);
    __builtin_prefetch(ap + 4, 0, 3);  // next k-chunk -> global_prefetch
    a.q[0] = ap[0]; a.q[1] = ap[1];
    const U128* wp = (const U128*)(Wt + (size_t)n * K + kc + lhalf * 16);
    bw.q[0] = wp[0]; bw.q[1] = wp[1];
    acc = wmma_bf16(a.v, bw.v, acc);
  }
  const float bb = bias ? bias[n] : 0.0f;
#pragma unroll
  for (int v = 0; v < 8; ++v) {
    const int r = mt * 16 + v + 8 * lhalf;
    float val = acc[v] + bb;
    if (do_relu) val = fmaxf(val, 0.0f);
    out[(size_t)r * N + nt * 16 + lrow] = f2bf(val);
  }
}

// ---------------------------------------------------------------------------
// One LSTM timestep: 6 stacked layers + 361-way logit head.
// Each wave owns 16 batch rows through all layers. Gates via WMMA
// (K=32, N=128 = 8 N-tiles; i/f/g/o live in acc tiles nt, nt+2, nt+4, nt+6).
// ---------------------------------------------------------------------------
__global__ void __launch_bounds__(256)
k_lstm_step(unsigned short* __restrict__ Zin, unsigned short* __restrict__ Hbuf,
            float* __restrict__ Cbuf,
            const unsigned short* __restrict__ WihT,
            const unsigned short* __restrict__ WhhT,
            const float* __restrict__ bih, const float* __restrict__ bhh,
            const unsigned short* __restrict__ W10t, const float* __restrict__ b10,
            float* __restrict__ logits) {
  __shared__ unsigned short stage[8 * 16 * 32];  // 16x32 bf16 per wave
  const int wave = threadIdx.x >> 5, lane = threadIdx.x & 31;
  const int lrow = lane & 15, lhalf = lane >> 4;
  const int rowbase = (blockIdx.x * 8 + wave) * 16;
  unsigned short* st = stage + wave * 512;

  BF16x16 aIn;  // current layer input, A-layout
  {
    const U128* p = (const U128*)(Zin + (size_t)(rowbase + lrow) * 32 + lhalf * 16);
    aIn.q[0] = p[0]; aIn.q[1] = p[1];
  }

  for (int l = 0; l < NLAY; ++l) {
    BF16x16 aH;  // previous-step hidden for this layer
    {
      const U128* p = (const U128*)(Hbuf + (size_t)l * B_SZ * 32 +
                                    (size_t)(rowbase + lrow) * 32 + lhalf * 16);
      aH.q[0] = p[0]; aH.q[1] = p[1];
    }
    v8f g[8];
#pragma unroll
    for (int nt = 0; nt < 8; ++nt) {
      BF16x16 bw, bu;
      const int n = nt * 16 + lrow;
      const U128* pw = (const U128*)(WihT + (size_t)l * 4096 + (size_t)n * 32 + lhalf * 16);
      bw.q[0] = pw[0]; bw.q[1] = pw[1];
      const U128* pu = (const U128*)(WhhT + (size_t)l * 4096 + (size_t)n * 32 + lhalf * 16);
      bu.q[0] = pu[0]; bu.q[1] = pu[1];
      v8f acc = {};
      acc = wmma_bf16(aIn.v, bw.v, acc);
      acc = wmma_bf16(aH.v, bu.v, acc);
      g[nt] = acc;
    }
#pragma unroll
    for (int nt = 0; nt < 2; ++nt) {
      const int col = nt * 16 + lrow;  // hidden index 0..31
      const float bi = bih[l * 128 + col]      + bhh[l * 128 + col];
      const float bf = bih[l * 128 + 32 + col] + bhh[l * 128 + 32 + col];
      const float bg = bih[l * 128 + 64 + col] + bhh[l * 128 + 64 + col];
      const float bo = bih[l * 128 + 96 + col] + bhh[l * 128 + 96 + col];
#pragma unroll
      for (int v = 0; v < 8; ++v) {
        const int r = rowbase + v + 8 * lhalf;
        const float si = sigmoidf_fast(g[nt][v] + bi);
        const float sf = sigmoidf_fast(g[nt + 2][v] + bf);
        const float tg = tanhf(g[nt + 4][v] + bg);
        const float so = sigmoidf_fast(g[nt + 6][v] + bo);
        const size_t ci = (size_t)l * B_SZ * 32 + (size_t)r * 32 + col;
        const float cn = sf * Cbuf[ci] + si * tg;
        Cbuf[ci] = cn;
        const float hn = so * tanhf(cn);
        const unsigned short hb = f2bf(hn);
        Hbuf[ci] = hb;
        st[(v + 8 * lhalf) * 32 + col] = hb;
        if (l == NLAY - 1) Zin[(size_t)r * 32 + col] = hb;  // next-step input
      }
    }
    __syncthreads();
    {
      const U128* p = (const U128*)(st + lrow * 32 + lhalf * 16);
      aIn.q[0] = p[0]; aIn.q[1] = p[1];
    }
    __syncthreads();
  }

  // logits = h5 @ fc10 + b10, N padded 361 -> 368 (23 tiles)
  for (int nt = 0; nt < 23; ++nt) {
    BF16x16 bw;
    const int n = nt * 16 + lrow;
    const U128* p = (const U128*)(W10t + (size_t)n * 32 + lhalf * 16);
    bw.q[0] = p[0]; bw.q[1] = p[1];
    v8f acc = {};
    acc = wmma_bf16(aIn.v, bw.v, acc);
    const int col = nt * 16 + lrow;
    const float bb = (col < VOCAB_N) ? b10[col] : -3.0e38f;
#pragma unroll
    for (int v = 0; v < 8; ++v) {
      const int r = rowbase + v + 8 * lhalf;
      logits[(size_t)r * VOCAB_P + col] = acc[v] + bb;
    }
  }
}

// ---------------------------------------------------------------------------
// Top-8 per row with jax.lax.top_k ordering (value desc, index asc for ties).
// ---------------------------------------------------------------------------
__global__ void k_topk(const float* __restrict__ logits, int* __restrict__ preds,
                       int t) {
  const int b = blockIdx.x * blockDim.x + threadIdx.x;
  if (b >= B_SZ) return;
  const float* lg = logits + (size_t)b * VOCAB_P;
  float pv = 3.4e38f;
  int pi = -1;
  for (int j = 0; j < KCURVES; ++j) {
    float bv = -3.4e38f;
    int bi = 0;
    for (int n = 0; n < VOCAB_N; ++n) {
      const float v = lg[n];
      const bool later = (v < pv) || (v == pv && n > pi);
      if (later && v > bv) { bv = v; bi = n; }
    }
    preds[(size_t)b * (TSTEPS * KCURVES) + t * KCURVES + j] = bi;
    pv = bv; pi = bi;
  }
}

// ---------------------------------------------------------------------------
extern "C" void kernel_launch(void* const* d_in, const int* in_sizes, int n_in,
                              void* d_out, int out_size, void* d_ws, size_t ws_size,
                              hipStream_t stream) {
  (void)in_sizes; (void)n_in; (void)out_size; (void)ws_size;
  const float* c_in   = (const float*)d_in[0];
  const float* x_in   = (const float*)d_in[1];
  const float* hidden = (const float*)d_in[2];
  const float* cell   = (const float*)d_in[3];
  const float* fc1_w  = (const float*)d_in[4];  const float* fc1_b  = (const float*)d_in[5];
  const float* fc2_w  = (const float*)d_in[6];  const float* fc2_b  = (const float*)d_in[7];
  const float* fc3_w  = (const float*)d_in[8];  const float* fc3_b  = (const float*)d_in[9];
  const float* fc4_w  = (const float*)d_in[10]; const float* fc4_b  = (const float*)d_in[11];
  const float* fc5_w  = (const float*)d_in[12]; const float* fc5_b  = (const float*)d_in[13];
  const float* fc6_w  = (const float*)d_in[14]; const float* fc6_b  = (const float*)d_in[15];
  const float* l_wih  = (const float*)d_in[16]; const float* l_whh  = (const float*)d_in[17];
  const float* l_bih  = (const float*)d_in[18]; const float* l_bhh  = (const float*)d_in[19];
  const float* fc10_w = (const float*)d_in[20]; const float* fc10_b = (const float*)d_in[21];
  int* preds = (int*)d_out;

  char* base = (char*)d_ws;
  size_t off = 0;
  auto alloc = [&](size_t bytes) -> void* {
    void* p = base + off;
    off = (off + bytes + 255) & ~(size_t)255;
    return p;
  };

  unsigned short* W1t  = (unsigned short*)alloc((size_t)256 * 64 * 2);
  unsigned short* W2t  = (unsigned short*)alloc((size_t)256 * 256 * 2);
  unsigned short* W3t  = (unsigned short*)alloc((size_t)512 * 320 * 2);
  unsigned short* W4t  = (unsigned short*)alloc((size_t)256 * 512 * 2);
  unsigned short* W5t  = (unsigned short*)alloc((size_t)128 * 256 * 2);
  unsigned short* W6t  = (unsigned short*)alloc((size_t)32 * 128 * 2);
  unsigned short* WihT = (unsigned short*)alloc((size_t)NLAY * 128 * 32 * 2);
  unsigned short* WhhT = (unsigned short*)alloc((size_t)NLAY * 128 * 32 * 2);
  unsigned short* W10t = (unsigned short*)alloc((size_t)VOCAB_P * 32 * 2);
  unsigned short* A3   = (unsigned short*)alloc((size_t)B_SZ * 320 * 2);
  unsigned short* A4   = (unsigned short*)alloc((size_t)B_SZ * 512 * 2);
  unsigned short* A5   = (unsigned short*)alloc((size_t)B_SZ * 256 * 2);
  unsigned short* A6   = (unsigned short*)alloc((size_t)B_SZ * 128 * 2);
  unsigned short* Zin  = (unsigned short*)alloc((size_t)B_SZ * 32 * 2);
  unsigned short* Hbuf = (unsigned short*)alloc((size_t)NLAY * B_SZ * 32 * 2);
  float*          Cbuf = (float*)alloc((size_t)NLAY * B_SZ * 32 * 4);
  float*        Logits = (float*)alloc((size_t)B_SZ * VOCAB_P * 4);

  // Weight prep (bf16, transposed, padded).
  k_transpose_bf16<<<64, 256, 0, stream>>>(fc1_w, W1t, 64, 256, 256, 1);
  k_transpose_bf16<<<256, 256, 0, stream>>>(fc2_w, W2t, 256, 256, 256, 1);
  k_transpose_bf16<<<640, 256, 0, stream>>>(fc3_w, W3t, 320, 512, 512, 1);
  k_transpose_bf16<<<512, 256, 0, stream>>>(fc4_w, W4t, 512, 256, 256, 1);
  k_transpose_bf16<<<128, 256, 0, stream>>>(fc5_w, W5t, 256, 128, 128, 1);
  k_transpose_bf16<<<16, 256, 0, stream>>>(fc6_w, W6t, 128, 32, 32, 1);
  k_transpose_bf16<<<96, 256, 0, stream>>>(l_wih, WihT, 32, 128, 128, NLAY);
  k_transpose_bf16<<<96, 256, 0, stream>>>(l_whh, WhhT, 32, 128, 128, NLAY);
  k_transpose_bf16<<<46, 256, 0, stream>>>(fc10_w, W10t, 32, VOCAB_N, VOCAB_P, 1);
  k_init_state<<<768, 256, 0, stream>>>(hidden, cell, Hbuf, Cbuf, NLAY * B_SZ * HIDN);

  // Particle encoder (async x staging + fc1 + fc2 + mean) -> A3 = [c | mean]
  k_particle_enc<<<B_SZ, 256, ENC_LDS_BYTES, stream>>>(x_in, c_in, W1t, fc1_b,
                                                       W2t, fc2_b, A3);

  // MLP head fc3..fc6 (tiles = (M/16)*(N/16), 8 tiles per block)
  k_gemm<<<(256 * 32) / 8, 256, 0, stream>>>(A3, W3t, fc3_b, A4, B_SZ, 320, 512, 1);
  k_gemm<<<(256 * 16) / 8, 256, 0, stream>>>(A4, W4t, fc4_b, A5, B_SZ, 512, 256, 1);
  k_gemm<<<(256 * 8) / 8, 256, 0, stream>>>(A5, W5t, fc5_b, A6, B_SZ, 256, 128, 1);
  k_gemm<<<(256 * 2) / 8, 256, 0, stream>>>(A6, W6t, fc6_b, Zin, B_SZ, 128, 32, 1);

  // 24 decode steps: LSTM stack + logits, then top-8 indices.
  for (int t = 0; t < TSTEPS; ++t) {
    k_lstm_step<<<B_SZ / 128, 256, 0, stream>>>(Zin, Hbuf, Cbuf, WihT, WhhT,
                                                l_bih, l_bhh, W10t, fc10_b, Logits);
    k_topk<<<B_SZ / 256, 256, 0, stream>>>(Logits, preds, t);
  }
}
